// ConvexWidthUpsampler_28922309771529
// MI455X (gfx1250) — matrix operand order repaired and checked by reading, not compiled, not run on hardware
//
#include <hip/hip_runtime.h>

// Fused CARAFE-style convex width-upsampler for gfx1250 (MI455X), wave32.
//   conv3x3(1->32) + PReLU + conv1x1(32->27) + softmax(9) + 3x3 blend, x3 width
// Strategy: each wave processes 32 pixels (one per lane). The two convs are
// executed as v_wmma_f32_16x16x32_f16 GEMMs (weights = A fragments built once
// per wave, activations = B fragments). Intermediate 32-channel activations
// never touch memory: they move accumulator->B-fragment within each lane,
// no cross-lane swap. Softmax + blend run per-lane in exact fp32.

typedef __attribute__((ext_vector_type(16))) _Float16 v16h;
typedef __attribute__((ext_vector_type(8)))  float    v8f;

#define Hd 512
#define Wd 512
#define Bd 16
#define NCH 32   // hidden channels
#define NMK 27   // 9 taps * 3 width phases

static __device__ __forceinline__ float shflx16(float v) {
    // swap the two 16-lane halves of the wave (wave32)
    return __shfl_xor(v, 16, 32);
}

__global__ __launch_bounds__(256) void caraf_fused_wmma(
    const float* __restrict__ x,
    const float* __restrict__ w1,     // [32][1][3][3]
    const float* __restrict__ b1,     // [32]
    const float* __restrict__ alpha,  // [32]
    const float* __restrict__ w2,     // [27][32][1][1]
    const float* __restrict__ b2,     // [27]
    float* __restrict__ out,          // [16][1][512][1536]
    int iters, int totalThreads)
{
    const int tid  = blockIdx.x * blockDim.x + threadIdx.x;
    const int lane = threadIdx.x & 31;
    const bool lo  = lane < 16;
    const int mrow = lane & 15;        // A/C row index within 16x16 tile
    const int klo  = lo ? 0 : 8;       // A-layout K base for this lane half

    // ---------------- per-wave constant A fragments (weights) ----------------
    // 16-bit A 16x32 layout: lane<16 holds K=0..7 (e=0..7) and K=16..23 (e=8..15)
    //                        lane>=16 holds K=8..15 and K=24..31.
    v16h A1[2] = {}, A2[2] = {};
    #pragma unroll
    for (int m = 0; m < 2; ++m) {
        const int c = mrow + 16 * m;              // output channel (A row)
        #pragma unroll
        for (int e = 0; e < 16; ++e) {
            const int K = e + klo + ((e >= 8) ? 8 : 0);
            const float v1 = (K < 9)   ? w1[c * 9 + K]  : 0.0f;   // 9 taps
            const float v2 = (c < NMK) ? w2[c * 32 + K] : 0.0f;   // 27 rows
            A1[m][e] = (_Float16)v1;
            A2[m][e] = (_Float16)v2;
        }
    }
    // bias / PReLU slope per C/D accumulator slot: row = 16m + r + (hi?8:0)
    float b1c[2][8], alc[2][8];
    #pragma unroll
    for (int m = 0; m < 2; ++m)
        #pragma unroll
        for (int r = 0; r < 8; ++r) {
            const int c = 16 * m + klo + r;
            b1c[m][r] = b1[c];
            alc[m][r] = alpha[c];
        }
    float b2c[NMK];
    #pragma unroll
    for (int c = 0; c < NMK; ++c) b2c[c] = b2[c];

    const v8f zacc = {};

    for (int it = 0; it < iters; ++it) {
        const int pix = tid + it * totalThreads;   // lane i of a wave -> pixel base+i
        const int w = pix & (Wd - 1);
        const int h = (pix >> 9) & (Hd - 1);
        const int b = pix >> 18;

        // -------- 3x3 zero-padded neighborhood of this lane's pixel (fp32) ----
        float nb[9];
        #pragma unroll
        for (int ky = 0; ky < 3; ++ky)
            #pragma unroll
            for (int kx = 0; kx < 3; ++kx) {
                const int hh = h + ky - 1, ww = w + kx - 1;
                const bool ok = (hh >= 0) && (hh < Hd) && (ww >= 0) && (ww < Wd);
                const int hc = min(max(hh, 0), Hd - 1);
                const int wc = min(max(ww, 0), Wd - 1);
                const float v = x[(b << 18) + (hc << 9) + wc];
                nb[ky * 3 + kx] = ok ? v : 0.0f;
            }

        // -------- B fragments for GEMM1 (taps x pixels), K=9 padded to 32 ----
        // B 32x16 layout (same as A): lanes 0-15 hold K=0..7,16..23, lanes 16-31
        // hold K=8..15,24..31. Taps 0..7 -> lo half e=0..7, tap 8 -> hi half e=0.
        v16h B1a = {}, B1b = {};
        #pragma unroll
        for (int k = 0; k < 8; ++k) {
            const float o = shflx16(nb[k]);                  // pixel lane^16
            B1a[k] = lo ? (_Float16)nb[k] : (_Float16)0.0f;  // pixels 0..15
            B1b[k] = lo ? (_Float16)o     : (_Float16)0.0f;  // pixels 16..31
        }
        {
            const float o8 = shflx16(nb[8]);
            if (!lo) { B1a[0] = (_Float16)o8; B1b[0] = (_Float16)nb[8]; }
        }

        // -------- GEMM1: h[32 x 32pix] = W1 x patches, + bias, PReLU ---------
        v8f hA[2][2];
        #pragma unroll
        for (int m = 0; m < 2; ++m) {
            hA[m][0] = __builtin_amdgcn_wmma_f32_16x16x32_f16(
                false, A1[m], false, B1a, (short)0, zacc, false, false);
            hA[m][1] = __builtin_amdgcn_wmma_f32_16x16x32_f16(
                false, A1[m], false, B1b, (short)0, zacc, false, false);
        }
        #pragma unroll
        for (int m = 0; m < 2; ++m)
            #pragma unroll
            for (int n = 0; n < 2; ++n)
                #pragma unroll
                for (int r = 0; r < 8; ++r) {
                    const float v = hA[m][n][r] + b1c[m][r];
                    hA[m][n][r] = (v > 0.0f) ? v : alc[m][r] * v;
                }

        // -------- repack h (C/D layout) into B fragments for GEMM2 -----------
        // lo lane needs ch 0..7,16..23 of its column; hi lane needs ch 8..15,24..31.
        v16h B2[2];
        #pragma unroll
        for (int n = 0; n < 2; ++n)
            #pragma unroll
            for (int r = 0; r < 8; ++r) {
                const float s0 = hA[0][n][r];       // ch  r      (lo) /  8+r (hi)
                const float s1 = hA[1][n][r];       // ch 16+r    (lo) / 24+r (hi)
                const float o0 = shflx16(s0);
                const float o1 = shflx16(s1);
                B2[n][r]     = (_Float16)s0;  // K=r / K=8+r
                B2[n][8 + r] = (_Float16)s1;  // K=16+r / 24+r
            }

        // -------- GEMM2: mask[27(->32) x 32pix] = W2 x h, K = 32 exact -------
        v8f mk[2][2];
        #pragma unroll
        for (int m = 0; m < 2; ++m) {
            mk[m][0] = __builtin_amdgcn_wmma_f32_16x16x32_f16(
                false, A2[m], false, B2[0], (short)0, zacc, false, false);
            mk[m][1] = __builtin_amdgcn_wmma_f32_16x16x32_f16(
                false, A2[m], false, B2[1], (short)0, zacc, false, false);
        }

        // -------- gather all 27 mask channels of THIS lane's pixel -----------
        float f[32];
        #pragma unroll
        for (int m = 0; m < 2; ++m)
            #pragma unroll
            for (int r = 0; r < 8; ++r) {
                const float sel = lo ? mk[m][0][r] : mk[m][1][r];
                const float oth = shflx16(sel);
                f[16 * m + r]     = lo ? sel : oth;
                f[16 * m + 8 + r] = lo ? oth : sel;
            }

        // -------- softmax over 9 taps (per width phase) + convex blend -------
        float res[3];
        #pragma unroll
        for (int v = 0; v < 3; ++v) {
            float mm[9], mx = -3.4e38f;
            #pragma unroll
            for (int k = 0; k < 9; ++k) {
                mm[k] = f[3 * k + v] + b2c[3 * k + v];
                mx = fmaxf(mx, mm[k]);
            }
            float s = 0.0f, acc = 0.0f;
            #pragma unroll
            for (int k = 0; k < 9; ++k) {
                const float e = __expf(mm[k] - mx);
                s   += e;
                acc += e * nb[k];
            }
            res[v] = acc / s;
        }

        float* o = out + ((size_t)b * Hd + h) * (size_t)(Wd * 3) + 3 * (size_t)w;
        o[0] = res[0];
        o[1] = res[1];
        o[2] = res[2];
    }
}

extern "C" void kernel_launch(void* const* d_in, const int* in_sizes, int n_in,
                              void* d_out, int out_size, void* d_ws, size_t ws_size,
                              hipStream_t stream) {
    // setup_inputs order: x, target_h, target_w, w1, b1, alpha, w2, b2
    const float* x     = (const float*)d_in[0];
    const float* w1    = (const float*)d_in[3];
    const float* b1    = (const float*)d_in[4];
    const float* alpha = (const float*)d_in[5];
    const float* w2    = (const float*)d_in[6];
    const float* b2    = (const float*)d_in[7];
    float* out = (float*)d_out;

    const int totalPix = Bd * Hd * Wd;         // 4,194,304
    const int threads  = 256;                  // 8 waves / block
    const int blocks   = 2048;
    const int totalThreads = threads * blocks; // 524,288
    const int iters = totalPix / totalThreads; // 8 (exact, no tail -> EXEC all-1s)

    caraf_fused_wmma<<<blocks, threads, 0, stream>>>(
        x, w1, b1, alpha, w2, b2, out, iters, totalThreads);
}
